// MSA_64330020159918
// MI455X (gfx1250) — matrix operand order repaired and checked by reading, not compiled, not run on hardware
//
#include <hip/hip_runtime.h>

typedef __attribute__((ext_vector_type(16))) __bf16 v16bf;
typedef __attribute__((ext_vector_type(8)))  __bf16 bf16x8;
typedef __attribute__((ext_vector_type(8)))  float  v8f;

#define NSEQ   1024
#define DMODEL 1024
#define NHEAD  16
#define HDIM   64
#define D3     3072
#define ATTN_SCALE 0.125f   // 1/sqrt(64)

// -------- CDNA5 async global->LDS copy (ASYNCcnt path), with safe fallback --------
#if defined(__has_builtin)
#if __has_builtin(__builtin_amdgcn_global_load_async_to_lds_b128)
#define HAVE_ASYNC_LDS 1
#endif
#endif
#ifndef HAVE_ASYNC_LDS
#define HAVE_ASYNC_LDS 0
#endif

// builtin prototype uses GCC-style int4 vector pointees:
//   void __builtin_amdgcn_global_load_async_to_lds_b128(v4i as1*, v4i as3*, imm, imm)
typedef int v4i_g __attribute__((vector_size(16)));
typedef __attribute__((address_space(1))) v4i_g* as1_v4i;
typedef __attribute__((address_space(3))) v4i_g* as3_v4i;

__device__ __forceinline__ void copy_b128_to_lds(const __bf16* g, __bf16* l) {
#if HAVE_ASYNC_LDS
  __builtin_amdgcn_global_load_async_to_lds_b128((as1_v4i)(void*)g,
                                                 (as3_v4i)(void*)l, 0, 0);
#else
  *(bf16x8*)l = *(const bf16x8*)g;
#endif
}

#if HAVE_ASYNC_LDS
#if __has_builtin(__builtin_amdgcn_s_wait_asynccnt)
#define WAIT_ASYNC(n) __builtin_amdgcn_s_wait_asynccnt(n)
#else
#define WAIT_ASYNC(n) asm volatile("s_wait_asynccnt %0" ::"n"(n) : "memory")
#endif
#else
#define WAIT_ASYNC(n) ((void)0)
#endif

// ---------------- WMMA fragment loaders (ISA 7.12.2 layouts) ----------------
// A fragment 16x32 bf16: lane(m=lane&15, h=lane>>4) holds row m,
// K = {8h..8h+7} U {16+8h..23+8h}  -> two contiguous b128 loads (row-major, ld elems).
__device__ __forceinline__ v16bf frag_a(const __bf16* base, int ld) {
  int lane = threadIdx.x & 31;
  int m = lane & 15, h = lane >> 4;
  bf16x8 lo = *(const bf16x8*)(base + m * ld + 8 * h);
  bf16x8 hi = *(const bf16x8*)(base + m * ld + 16 + 8 * h);
  v16bf a;
#pragma unroll
  for (int i = 0; i < 8; ++i) { a[i] = lo[i]; a[i + 8] = hi[i]; }
  return a;
}

// B fragment 32x16 bf16: lane(n=lane&15, h=lane>>4) holds column n,
// K = {8h..8h+7} U {16+8h..23+8h} -> two b128 loads from column-major storage:
// element (k, n) at base[n*ld + k].
__device__ __forceinline__ v16bf frag_b(const __bf16* base, int ld) {
  int lane = threadIdx.x & 31;
  int n = lane & 15, h = lane >> 4;
  bf16x8 lo = *(const bf16x8*)(base + n * ld + 8 * h);
  bf16x8 hi = *(const bf16x8*)(base + n * ld + 16 + 8 * h);
  v16bf b;
#pragma unroll
  for (int i = 0; i < 8; ++i) { b[i] = lo[i]; b[i + 8] = hi[i]; }
  return b;
}

#define WMMA_BF16(A, B, C) \
  __builtin_amdgcn_wmma_f32_16x16x32_bf16(false, (A), false, (B), (short)0, (C), false, false)

// half-wave (16-lane) reductions: score row m=e+8h lives in one 16-lane half
__device__ __forceinline__ float half_max16(float v) {
#pragma unroll
  for (int m = 1; m < 16; m <<= 1) v = fmaxf(v, __shfl_xor(v, m, 32));
  return v;
}
__device__ __forceinline__ float half_sum16(float v) {
#pragma unroll
  for (int m = 1; m < 16; m <<= 1) v += __shfl_xor(v, m, 32);
  return v;
}

// ---------------- LayerNorm + cast to bf16 ----------------
__global__ __launch_bounds__(256) void ln_cast_kernel(const float* __restrict__ z,
                                                      const float* __restrict__ g,
                                                      const float* __restrict__ bta,
                                                      __bf16* __restrict__ xb) {
  __shared__ float s1[256], s2[256];
  int r = blockIdx.x, t = threadIdx.x;
  const float* zr = z + (size_t)r * DMODEL;
  float v[4], sum = 0.f, sq = 0.f;
#pragma unroll
  for (int i = 0; i < 4; ++i) {
    float x = zr[t + 256 * i];
    v[i] = x; sum += x; sq += x * x;
  }
  s1[t] = sum; s2[t] = sq;
  __syncthreads();
  for (int o = 128; o > 0; o >>= 1) {
    if (t < o) { s1[t] += s1[t + o]; s2[t] += s2[t + o]; }
    __syncthreads();
  }
  float mu  = s1[0] * (1.0f / DMODEL);
  float var = s2[0] * (1.0f / DMODEL) - mu * mu;
  float rstd = rsqrtf(var + 1e-5f);
#pragma unroll
  for (int i = 0; i < 4; ++i) {
    int e = t + 256 * i;
    xb[(size_t)r * DMODEL + e] = (__bf16)((v[i] - mu) * rstd * g[e] + bta[e]);
  }
}

// -------- cast + transpose weights via 32x32 LDS tile (coalesced both sides) --------
// src [rows, cols] fp32 -> dst [cols, rows] bf16
__global__ __launch_bounds__(256) void cast_transpose_kernel(const float* __restrict__ src,
                                                             __bf16* __restrict__ dst,
                                                             int rows, int cols) {
  __shared__ float tile[32][33];
  int bx = blockIdx.x * 32;          // original col base
  int by = blockIdx.y * 32;          // original row base
  int tx = threadIdx.x & 31, ty = threadIdx.x >> 5;   // 8 rows per pass
#pragma unroll
  for (int i = 0; i < 4; ++i) {
    int r = ty + i * 8;
    tile[r][tx] = src[(size_t)(by + r) * cols + bx + tx];
  }
  __syncthreads();
#pragma unroll
  for (int i = 0; i < 4; ++i) {
    int r = ty + i * 8;              // row in transposed output = original col
    dst[(size_t)(bx + r) * rows + by + tx] = (__bf16)tile[tx][r];
  }
}

// ======== shared GEMM core: 128x64 block tile, K-step 64, double-buffered LDS ========
// As: row-major ld=64 (16KB/buf).  Bs: column-major (k,n)->Bs[n*64+k] (8KB/buf).
__device__ __forceinline__ void gemm_issue_tile(const __bf16* __restrict__ X,
                                                const __bf16* __restrict__ WT,
                                                __bf16* Asb, __bf16* Bsb,
                                                int m0, int n0, int k0) {
#pragma unroll
  for (int rr = 0; rr < 4; ++rr) {           // A: 1024 b128 chunks / 256 thr = 4
    int c = threadIdx.x + rr * 256;
    int row = c >> 3, seg = c & 7;
    copy_b128_to_lds(X + (size_t)(m0 + row) * DMODEL + k0 + seg * 8,
                     Asb + row * 64 + seg * 8);
  }
#pragma unroll
  for (int rr = 0; rr < 2; ++rr) {           // B: 512 b128 chunks / 256 thr = 2
    int c = threadIdx.x + rr * 256;
    int col = c >> 3, seg = c & 7;
    copy_b128_to_lds(WT + (size_t)(n0 + col) * DMODEL + k0 + seg * 8,
                     Bsb + col * 64 + seg * 8);
  }
}

__device__ __forceinline__ void gemm_compute(const __bf16* Asb, const __bf16* Bsb,
                                             int wm, int wn, v8f acc[2][2]) {
#pragma unroll
  for (int kk = 0; kk < 64; kk += 32) {
    v16bf a0 = frag_a(Asb + (wm * 32) * 64 + kk, 64);
    v16bf a1 = frag_a(Asb + (wm * 32 + 16) * 64 + kk, 64);
    v16bf b0 = frag_b(Bsb + (wn * 32) * 64 + kk, 64);
    v16bf b1 = frag_b(Bsb + (wn * 32 + 16) * 64 + kk, 64);
    acc[0][0] = WMMA_BF16(a0, b0, acc[0][0]);
    acc[0][1] = WMMA_BF16(a0, b1, acc[0][1]);
    acc[1][0] = WMMA_BF16(a1, b0, acc[1][0]);
    acc[1][1] = WMMA_BF16(a1, b1, acc[1][1]);
  }
}

#define GEMM_PIPELINE(X, WT)                                                     \
  __shared__ __bf16 As[2][128 * 64];                                             \
  __shared__ __bf16 Bs[2][64 * 64];                                              \
  int m0 = blockIdx.y * 128, n0 = blockIdx.x * 64;                               \
  int w = threadIdx.x >> 5, lane = threadIdx.x & 31;                             \
  int wm = w & 3, wn = w >> 2;                                                   \
  v8f acc[2][2];                                                                 \
  _Pragma("unroll") for (int i = 0; i < 2; ++i)                                  \
      _Pragma("unroll") for (int j = 0; j < 2; ++j)                              \
          _Pragma("unroll") for (int e = 0; e < 8; ++e) acc[i][j][e] = 0.f;      \
  gemm_issue_tile(X, WT, As[0], Bs[0], m0, n0, 0);                               \
  for (int k0 = 0; k0 < DMODEL; k0 += 64) {                                      \
    int cur = (k0 >> 6) & 1;                                                     \
    if (k0 + 64 < DMODEL) {                                                      \
      gemm_issue_tile(X, WT, As[cur ^ 1], Bs[cur ^ 1], m0, n0, k0 + 64);         \
      WAIT_ASYNC(6); /* 6 newest ops = next tile; older (current) complete */    \
    } else {                                                                     \
      WAIT_ASYNC(0);                                                             \
    }                                                                            \
    __syncthreads();                                                             \
    gemm_compute(As[cur], Bs[cur], wm, wn, acc);                                 \
    __syncthreads();                                                             \
  }

// ---------------- QKV GEMM: [4096,1024] x [1024,3072] -> scatter Q,K,V bf16 ----------------
__global__ __launch_bounds__(256) void gemm_qkv_kernel(const __bf16* __restrict__ X,
                                                       const __bf16* __restrict__ WT,
                                                       __bf16* __restrict__ Qd,
                                                       __bf16* __restrict__ Kd,
                                                       __bf16* __restrict__ Vd) {
  GEMM_PIPELINE(X, WT)
  // epilogue: scatter into head layouts. Q,K: [b,h,n,hd]; V: [b,h,hd,n]
  int ncol = lane & 15, hh = lane >> 4;
#pragma unroll
  for (int i = 0; i < 2; ++i)
#pragma unroll
    for (int j = 0; j < 2; ++j)
#pragma unroll
      for (int e = 0; e < 8; ++e) {
        int row = m0 + wm * 32 + i * 16 + e + 8 * hh;
        int col = n0 + wn * 32 + j * 16 + ncol;
        int bb = row >> 10, n = row & 1023;
        int which = col >> 10, dc = col & 1023;
        int hg = dc >> 6, hd = dc & 63;
        __bf16 val = (__bf16)acc[i][j][e];
        if (which == 0)
          Qd[(((size_t)bb * NHEAD + hg) * NSEQ + n) * HDIM + hd] = val;
        else if (which == 1)
          Kd[(((size_t)bb * NHEAD + hg) * NSEQ + n) * HDIM + hd] = val;
        else
          Vd[(((size_t)bb * NHEAD + hg) * HDIM + hd) * NSEQ + n] = val;
      }
}

// ---------------- projection GEMM + bias + residual -> fp32 out ----------------
__global__ __launch_bounds__(256) void gemm_proj_kernel(const __bf16* __restrict__ X,
                                                        const __bf16* __restrict__ WT,
                                                        const float* __restrict__ bias,
                                                        const float* __restrict__ zres,
                                                        float* __restrict__ out) {
  GEMM_PIPELINE(X, WT)
  int ncol = lane & 15, hh = lane >> 4;
#pragma unroll
  for (int i = 0; i < 2; ++i)
#pragma unroll
    for (int j = 0; j < 2; ++j)
#pragma unroll
      for (int e = 0; e < 8; ++e) {
        int row = m0 + wm * 32 + i * 16 + e + 8 * hh;
        int col = n0 + wn * 32 + j * 16 + ncol;
        size_t idx = (size_t)row * DMODEL + col;
        out[idx] = zres[idx] + acc[i][j][e] + bias[col];
      }
}

// ---------------- attention: flash-style, per-wave 16-query tile ----------------
__global__ __launch_bounds__(128) void attn_kernel(const __bf16* __restrict__ Qd,
                                                   const __bf16* __restrict__ Kd,
                                                   const __bf16* __restrict__ Vd,
                                                   __bf16* __restrict__ msa) {
  __shared__ __bf16 Ps[4 * 16 * 32];          // per-wave P tile, row-major ld=32
  int bh = blockIdx.y;                        // 0..63 == b*16 + h
  int b = bh >> 4, h = bh & 15;
  int w = threadIdx.x >> 5, lane = threadIdx.x & 31;
  int ncol = lane & 15, hh = lane >> 4;
  const __bf16* Qb = Qd + (size_t)bh * NSEQ * HDIM;
  const __bf16* Kb = Kd + (size_t)bh * NSEQ * HDIM;
  const __bf16* Vb = Vd + (size_t)bh * HDIM * NSEQ;
  __bf16* Pw = Ps + w * (16 * 32);
  int q0 = blockIdx.x * 64 + w * 16;

  v16bf qa0 = frag_a(Qb + (size_t)q0 * HDIM, HDIM);        // hd 0..31
  v16bf qa1 = frag_a(Qb + (size_t)q0 * HDIM + 32, HDIM);   // hd 32..63

  float rmax[8], rsum[8];
  v8f o[4];
#pragma unroll
  for (int e = 0; e < 8; ++e) { rmax[e] = -1e30f; rsum[e] = 0.f; }
#pragma unroll
  for (int j = 0; j < 4; ++j)
#pragma unroll
    for (int e = 0; e < 8; ++e) o[j][e] = 0.f;

  for (int kt = 0; kt < NSEQ; kt += 32) {
    v8f s0, s1;
#pragma unroll
    for (int e = 0; e < 8; ++e) { s0[e] = 0.f; s1[e] = 0.f; }
    // S = Q K^T for 32 keys: K rows are contiguous over hd -> direct global B-frags
    s0 = WMMA_BF16(qa0, frag_b(Kb + (size_t)kt * HDIM, HDIM), s0);
    s0 = WMMA_BF16(qa1, frag_b(Kb + (size_t)kt * HDIM + 32, HDIM), s0);
    s1 = WMMA_BF16(qa0, frag_b(Kb + (size_t)(kt + 16) * HDIM, HDIM), s1);
    s1 = WMMA_BF16(qa1, frag_b(Kb + (size_t)(kt + 16) * HDIM + 32, HDIM), s1);
    // online softmax per score row (m = e + 8*hh), reduce across the 16-lane half
#pragma unroll
    for (int e = 0; e < 8; ++e) {
      float a = s0[e] * ATTN_SCALE;
      float c = s1[e] * ATTN_SCALE;
      float tmax = half_max16(fmaxf(a, c));
      float nm = fmaxf(rmax[e], tmax);
      float alpha = __expf(rmax[e] - nm);
      float p0 = __expf(a - nm);
      float p1 = __expf(c - nm);
      rsum[e] = rsum[e] * alpha + half_sum16(p0 + p1);
      rmax[e] = nm;
#pragma unroll
      for (int j = 0; j < 4; ++j) o[j][e] *= alpha;
      Pw[(e + 8 * hh) * 32 + ncol]      = (__bf16)p0;
      Pw[(e + 8 * hh) * 32 + 16 + ncol] = (__bf16)p1;
    }
    // O += P V : V is [hd, n] so columns (hd) are contiguous over keys
    v16bf pa = frag_a(Pw, 32);
#pragma unroll
    for (int j = 0; j < 4; ++j) {
      v16bf vb = frag_b(Vb + (size_t)(j * 16) * NSEQ + kt, NSEQ);
      o[j] = WMMA_BF16(pa, vb, o[j]);
    }
  }
  // epilogue: normalize, write msa [b, n, D] at column h*64 + hd
  __bf16* outb = msa + (size_t)b * NSEQ * DMODEL;
#pragma unroll
  for (int e = 0; e < 8; ++e) {
    float inv = 1.0f / rsum[e];
    int n = q0 + e + 8 * hh;
#pragma unroll
    for (int j = 0; j < 4; ++j)
      outb[(size_t)n * DMODEL + h * HDIM + j * 16 + ncol] = (__bf16)(o[j][e] * inv);
  }
}

// ---------------- launch ----------------
extern "C" void kernel_launch(void* const* d_in, const int* in_sizes, int n_in,
                              void* d_out, int out_size, void* d_ws, size_t ws_size,
                              hipStream_t stream) {
  const float* z      = (const float*)d_in[0];
  const float* gamma  = (const float*)d_in[1];
  const float* beta   = (const float*)d_in[2];
  const float* w_qkv  = (const float*)d_in[3];
  const float* w_proj = (const float*)d_in[4];
  const float* b_proj = (const float*)d_in[5];
  float* out = (float*)d_out;

  char* ws = (char*)d_ws;
  __bf16* xb     = (__bf16*)(ws);                          // 4096x1024      (8 MiB)
  __bf16* wqkvT  = (__bf16*)(ws + ((size_t)8  << 20));     // [3072,1024]    (6 MiB)
  __bf16* wprojT = (__bf16*)(ws + ((size_t)14 << 20));     // [1024,1024]    (2 MiB)
  __bf16* Qh     = (__bf16*)(ws + ((size_t)16 << 20));     // [b,h,n,hd]     (8 MiB)
  __bf16* Kh     = (__bf16*)(ws + ((size_t)24 << 20));     // [b,h,n,hd]     (8 MiB)
  __bf16* Vh     = (__bf16*)(ws + ((size_t)32 << 20));     // [b,h,hd,n]     (8 MiB)
  __bf16* msa    = (__bf16*)(ws + ((size_t)40 << 20));     // [b,n,D]        (8 MiB)

  ln_cast_kernel<<<4096, 256, 0, stream>>>(z, gamma, beta, xb);
  cast_transpose_kernel<<<dim3(D3 / 32, DMODEL / 32), 256, 0, stream>>>(w_qkv, wqkvT, DMODEL, D3);
  cast_transpose_kernel<<<dim3(DMODEL / 32, DMODEL / 32), 256, 0, stream>>>(w_proj, wprojT, DMODEL, DMODEL);
  gemm_qkv_kernel<<<dim3(D3 / 64, 4096 / 128), 256, 0, stream>>>(xb, wqkvT, Qh, Kh, Vh);
  attn_kernel<<<dim3(NSEQ / 64, 4 * NHEAD), 128, 0, stream>>>(Qh, Kh, Vh, msa);
  gemm_proj_kernel<<<dim3(DMODEL / 64, 4096 / 128), 256, 0, stream>>>(msa, wprojT, b_proj, z, out);
}
